// C2IBKE_25589415149697
// MI455X (gfx1250) — matrix operand order repaired and checked by reading, not compiled, not run on hardware
//
#include <hip/hip_runtime.h>
#include <hip/hip_bf16.h>
#include <math.h>

// Problem constants (match reference)
#define N_ENTS 50000
#define DIMF   200
#define BATCHB 1024
#define KP1    224   // 200 padded to 7*32
#define KP2    416   // 400 padded to 13*32

typedef _Float16 v16h __attribute__((ext_vector_type(16)));
typedef _Float16 v8h  __attribute__((ext_vector_type(8)));
typedef _Float16 v4h  __attribute__((ext_vector_type(4)));
typedef float    v8f  __attribute__((ext_vector_type(8)));

// ---------------- WMMA helpers (wave32, 16x16x32 f16 -> f32) ----------------

__device__ __forceinline__ v8f wmma_f16(v16h a, v16h b, v8f c) {
  // (neg_a, A, neg_b, B, c_mod, C, reuse_a, reuse_b)
  return __builtin_amdgcn_wmma_f32_16x16x32_f16(false, a, false, b, (short)0, c,
                                                false, false);
}

// A fragment: LDS tile stored [16 rows][stride K halves], row m, K-base kb.
// Lane layout (ISA 7.12.2, 16-bit A 16x32): lanes0-15: e0..7->K=kb..kb+7,
// e8..15->K=kb+16..23 ; lanes16-31: +8.
__device__ __forceinline__ v16h frag_a(const _Float16* As, int stride, int m,
                                       int kb, int hi) {
  v8h lo = *(const v8h*)(As + m * stride + kb + hi * 8);
  v8h hh = *(const v8h*)(As + m * stride + kb + 16 + hi * 8);
  return __builtin_shufflevector(lo, hh, 0, 1, 2, 3, 4, 5, 6, 7, 8, 9, 10, 11,
                                 12, 13, 14, 15);
}

// B fragment: LDS tile stored N-major [16 n][stride K halves] (i.e. B^T rows).
// Lane n = lane&15; lanes0-15: e0..7->K=kb..kb+7, e8..15->K=kb+16..23 ; lanes16-31: +8.
__device__ __forceinline__ v16h frag_b(const _Float16* Bs, int stride, int n,
                                       int kb, int hi) {
  const v8h* p0 = (const v8h*)(Bs + n * stride + kb + hi * 8);
  v8h lo = p0[0];
  v8h hh = p0[2];
  return __builtin_shufflevector(lo, hh, 0, 1, 2, 3, 4, 5, 6, 7, 8, 9, 10, 11,
                                 12, 13, 14, 15);
}

__device__ __forceinline__ float concrete_gate(float logit, float u) {
  const float bias = 1e-4f;
  float eps = (2.0f * bias - 1.0f) * u + (1.0f - bias);
  float noise = logf(eps) - log1pf(-eps);
  float z = 2.0f * (noise + logit);
  return 1.0f / (1.0f + __expf(-z));
}

// ---------------- Node gate: mask[i][n], i = blockIdx.y ----------------
// h = relu(e0 @ W1[i] + b1[i]); logit = h @ W2[i] + b2[i]; mask = gate(logit, eps)
__global__ __launch_bounds__(128) void node_gate_kernel(
    const float* __restrict__ e0, const float* __restrict__ W1all,
    const float* __restrict__ b1all, const float* __restrict__ W2all,
    const float* __restrict__ b2all, const float* __restrict__ epsall,
    float* __restrict__ maskall) {
  const int gi = blockIdx.y;
  const float* W1 = W1all + (size_t)gi * DIMF * DIMF;
  const float* b1 = b1all + gi * DIMF;
  const float* W2 = W2all + gi * DIMF;
  const float b2 = b2all[gi];
  const float* eps = epsall + (size_t)gi * N_ENTS;
  float* mask = maskall + (size_t)gi * N_ENTS;

  __shared__ _Float16 As[64 * KP1];
  __shared__ _Float16 Bs[16 * KP1];
  const int tid = threadIdx.x;
  const int lane = tid & 31, wave = tid >> 5;
  const int hi = lane >> 4, l16 = lane & 15;
  const int rowBase = blockIdx.x * 64;

  // stage A: 64 entity rows, f32 -> f16, K padded with zeros
  for (int c = tid; c < 64 * (KP1 / 4); c += 128) {
    int m = c / (KP1 / 4);
    int kq = (c % (KP1 / 4)) * 4;
    int row = rowBase + m;
    v4h h4 = {};
    if (row < N_ENTS && kq < DIMF) {
      float4 f = *(const float4*)(e0 + (size_t)row * DIMF + kq);
      h4[0] = (_Float16)f.x; h4[1] = (_Float16)f.y;
      h4[2] = (_Float16)f.z; h4[3] = (_Float16)f.w;
    }
    *(v4h*)(As + m * KP1 + kq) = h4;
  }
  __syncthreads();

  // cache the wave's entire A K-strip in registers (invariant over N tiles)
  v16h afrag[7];
#pragma unroll
  for (int kt = 0; kt < 7; ++kt)
    afrag[kt] = frag_a(As + wave * 16 * KP1, KP1, l16, kt * 32, hi);

  float rowsum[8];
#pragma unroll
  for (int r = 0; r < 8; ++r) rowsum[r] = 0.f;

  for (int nt = 0; nt < 13; ++nt) {
    const int nb = nt * 16;
    __syncthreads();
    // stage B^T: Bs[n][k] = W1[k][nb+n]
    for (int c = tid; c < 16 * (KP1 / 4); c += 128) {
      int n = c & 15;
      int kq = (c >> 4) * 4;
      int coln = nb + n;
      v4h h4 = {};
      if (coln < DIMF && kq < DIMF) {
#pragma unroll
        for (int j = 0; j < 4; ++j)
          h4[j] = (_Float16)W1[(size_t)(kq + j) * DIMF + coln];
      }
      *(v4h*)(Bs + n * KP1 + kq) = h4;
    }
    __syncthreads();
    v8f acc = {};
#pragma unroll
    for (int kt = 0; kt < 7; ++kt) {
      v16h b = frag_b(Bs, KP1, l16, kt * 32, hi);
      acc = wmma_f16(afrag[kt], b, acc);
    }
    int col = nb + l16;
    float b1v = (col < DIMF) ? b1[col] : 0.f;
    float w2v = (col < DIMF) ? W2[col] : 0.f;
#pragma unroll
    for (int r = 0; r < 8; ++r) {
      float h = acc[r] + b1v;
      h = h > 0.f ? h : 0.f;
      rowsum[r] += h * w2v;
    }
  }
  // reduce over the 16 lanes of each half (C layout: half0 rows r, half1 rows r+8)
#pragma unroll
  for (int r = 0; r < 8; ++r) {
    rowsum[r] += __shfl_xor(rowsum[r], 1, 32);
    rowsum[r] += __shfl_xor(rowsum[r], 2, 32);
    rowsum[r] += __shfl_xor(rowsum[r], 4, 32);
    rowsum[r] += __shfl_xor(rowsum[r], 8, 32);
  }
  if (l16 == 0) {
#pragma unroll
    for (int r = 0; r < 8; ++r) {
      int row = rowBase + wave * 16 + hi * 8 + r;
      if (row < N_ENTS) mask[row] = concrete_gate(rowsum[r] + b2, eps[row]);
    }
  }
}

// ---------------- Edge gate (only i=1 is used by the reference) ----------------
// he = relu(concat(e0[row],e0[col]) @ W1 + b1); mask = gate(he @ W2 + b2, eps)
__global__ __launch_bounds__(64) void edge_gate_kernel(
    const float* __restrict__ e0, const int* __restrict__ arow,
    const int* __restrict__ acol, const float* __restrict__ W1,
    const float* __restrict__ b1, const float* __restrict__ W2,
    const float* __restrict__ b2p, const float* __restrict__ eps,
    float* __restrict__ emask, int E) {
  __shared__ _Float16 As[32 * KP2];
  __shared__ _Float16 Bs[16 * KP2];
  const int tid = threadIdx.x;
  const int lane = tid & 31, wave = tid >> 5;
  const int hi = lane >> 4, l16 = lane & 15;
  const int edgeBase = blockIdx.x * 32;
  const float b2 = b2p[0];

  for (int c = tid; c < 32 * (KP2 / 4); c += 64) {
    int m = c / (KP2 / 4);
    int kq = (c % (KP2 / 4)) * 4;
    int eid = edgeBase + m;
    v4h h4 = {};
    if (eid < E && kq < 2 * DIMF) {
      int src = (kq < DIMF) ? arow[eid] : acol[eid];
      int k0 = (kq < DIMF) ? kq : (kq - DIMF);
      float4 f = *(const float4*)(e0 + (size_t)src * DIMF + k0);
      h4[0] = (_Float16)f.x; h4[1] = (_Float16)f.y;
      h4[2] = (_Float16)f.z; h4[3] = (_Float16)f.w;
    }
    *(v4h*)(As + m * KP2 + kq) = h4;
  }
  __syncthreads();

  v16h afrag[13];
#pragma unroll
  for (int kt = 0; kt < 13; ++kt)
    afrag[kt] = frag_a(As + wave * 16 * KP2, KP2, l16, kt * 32, hi);

  float rowsum[8];
#pragma unroll
  for (int r = 0; r < 8; ++r) rowsum[r] = 0.f;

  for (int nt = 0; nt < 13; ++nt) {
    const int nb = nt * 16;
    __syncthreads();
    for (int c = tid; c < 16 * (KP2 / 4); c += 64) {
      int n = c & 15;
      int kq = (c >> 4) * 4;
      int coln = nb + n;
      v4h h4 = {};
      if (coln < DIMF && kq < 2 * DIMF) {
#pragma unroll
        for (int j = 0; j < 4; ++j)
          h4[j] = (_Float16)W1[(size_t)(kq + j) * DIMF + coln];
      }
      *(v4h*)(Bs + n * KP2 + kq) = h4;
    }
    __syncthreads();
    v8f acc = {};
#pragma unroll
    for (int kt = 0; kt < 13; ++kt) {
      v16h b = frag_b(Bs, KP2, l16, kt * 32, hi);
      acc = wmma_f16(afrag[kt], b, acc);
    }
    int col = nb + l16;
    float b1v = (col < DIMF) ? b1[col] : 0.f;
    float w2v = (col < DIMF) ? W2[col] : 0.f;
#pragma unroll
    for (int r = 0; r < 8; ++r) {
      float h = acc[r] + b1v;
      h = h > 0.f ? h : 0.f;
      rowsum[r] += h * w2v;
    }
  }
#pragma unroll
  for (int r = 0; r < 8; ++r) {
    rowsum[r] += __shfl_xor(rowsum[r], 1, 32);
    rowsum[r] += __shfl_xor(rowsum[r], 2, 32);
    rowsum[r] += __shfl_xor(rowsum[r], 4, 32);
    rowsum[r] += __shfl_xor(rowsum[r], 8, 32);
  }
  if (l16 == 0) {
#pragma unroll
    for (int r = 0; r < 8; ++r) {
      int eid = edgeBase + wave * 16 + hi * 8 + r;
      if (eid < E) emask[eid] = concrete_gate(rowsum[r] + b2, eps[eid]);
    }
  }
}

// ---------------- RESCAL per-sample contraction: hr[b] = lhs[b] @ rel1[b] ----------------
__global__ __launch_bounds__(256) void self_hr_kernel(
    const float* __restrict__ e0, const float* __restrict__ rel_emb,
    const int* __restrict__ sub, const int* __restrict__ rel,
    float* __restrict__ hr) {
  const int b = blockIdx.x;
  __shared__ float lhs[DIMF];
  const float* R = rel_emb + (size_t)rel[b] * DIMF * DIMF;
  const float* L = e0 + (size_t)sub[b] * DIMF;
  const int t = threadIdx.x;
  if (t < DIMF) lhs[t] = L[t];
  __syncthreads();
  if (t < DIMF) {
    float s = 0.f;
    for (int d = 0; d < DIMF; ++d) s += lhs[d] * R[(size_t)d * DIMF + t];
    hr[(size_t)b * DIMF + t] = s;
  }
}

// ---------------- scores = sigmoid(hr @ e0^T) : M=1024, N=50000, K=200 ----------------
__global__ __launch_bounds__(128) void scores_kernel(
    const float* __restrict__ hr, const float* __restrict__ e0,
    float* __restrict__ out) {
  __shared__ _Float16 As[64 * KP1];
  __shared__ _Float16 Bs[16 * KP1];
  const int tid = threadIdx.x;
  const int lane = tid & 31, wave = tid >> 5;
  const int hi = lane >> 4, l16 = lane & 15;
  const int rowBase = blockIdx.y * 64;
  const int colChunk = blockIdx.x * 256;

  for (int c = tid; c < 64 * (KP1 / 4); c += 128) {
    int m = c / (KP1 / 4);
    int kq = (c % (KP1 / 4)) * 4;
    v4h h4 = {};
    if (kq < DIMF) {
      float4 f = *(const float4*)(hr + (size_t)(rowBase + m) * DIMF + kq);
      h4[0] = (_Float16)f.x; h4[1] = (_Float16)f.y;
      h4[2] = (_Float16)f.z; h4[3] = (_Float16)f.w;
    }
    *(v4h*)(As + m * KP1 + kq) = h4;
  }
  __syncthreads();

  v16h afrag[7];
#pragma unroll
  for (int kt = 0; kt < 7; ++kt)
    afrag[kt] = frag_a(As + wave * 16 * KP1, KP1, l16, kt * 32, hi);

  for (int nt = 0; nt < 16; ++nt) {
    const int nb = colChunk + nt * 16;
    __syncthreads();
    // B^T rows are simply e0 rows (scores[b,n] = hr[b] . e0[n])
    for (int c = tid; c < 16 * (KP1 / 4); c += 128) {
      int n = c & 15;
      int kq = (c >> 4) * 4;
      int ent = nb + n;
      v4h h4 = {};
      if (ent < N_ENTS && kq < DIMF) {
        float4 f = *(const float4*)(e0 + (size_t)ent * DIMF + kq);
        h4[0] = (_Float16)f.x; h4[1] = (_Float16)f.y;
        h4[2] = (_Float16)f.z; h4[3] = (_Float16)f.w;
      }
      *(v4h*)(Bs + n * KP1 + kq) = h4;
    }
    __syncthreads();
    v8f acc = {};
#pragma unroll
    for (int kt = 0; kt < 7; ++kt) {
      v16h b = frag_b(Bs, KP1, l16, kt * 32, hi);
      acc = wmma_f16(afrag[kt], b, acc);
    }
    int col = nb + l16;
    if (col < N_ENTS) {
#pragma unroll
      for (int r = 0; r < 8; ++r) {
        int row = rowBase + wave * 16 + hi * 8 + r;
        out[(size_t)row * N_ENTS + col] = 1.0f / (1.0f + __expf(-acc[r]));
      }
    }
  }
}

// ---------------- CSR rowptr from sorted COO row array ----------------
__global__ void rowptr_kernel(const int* __restrict__ row, int E, int N,
                              int* __restrict__ rowptr) {
  int r = blockIdx.x * blockDim.x + threadIdx.x;
  if (r > N) return;
  int lo = 0, hiid = E;
  while (lo < hiid) {
    int mid = (lo + hiid) >> 1;
    if (row[mid] < r) lo = mid + 1; else hiid = mid;
  }
  rowptr[r] = lo;
}

// ---------------- SpMM: y = (COO with optional per-edge multiplier) @ x ----------------
__global__ __launch_bounds__(256) void spmm_kernel(
    const int* __restrict__ rowptr, const int* __restrict__ col,
    const float* __restrict__ val, const float* __restrict__ vmul,
    const float* __restrict__ x, float* __restrict__ y) {
  const int r = blockIdx.x;
  const int f = threadIdx.x;
  if (f >= DIMF) return;
  const int s = rowptr[r], e = rowptr[r + 1];
  float acc = 0.f;
  for (int i = s; i < e; ++i) {
    if (i + 1 < e) __builtin_prefetch(x + (size_t)col[i + 1] * DIMF + f, 0, 0);
    float v = val[i];
    if (vmul) v *= vmul[i];
    acc += v * x[(size_t)col[i] * DIMF + f];
  }
  y[(size_t)r * DIMF + f] = acc;
}

// ---------------- elementwise mix: y = m*xa + (1-m)*xb (m per row) ----------------
__global__ __launch_bounds__(256) void combine_kernel(
    const float* __restrict__ mask, const float* __restrict__ xa,
    const float* __restrict__ xb, float* __restrict__ y) {
  size_t idx = (size_t)blockIdx.x * blockDim.x + threadIdx.x;
  if (idx >= (size_t)N_ENTS * DIMF) return;
  float m = mask[idx / DIMF];
  y[idx] = m * xa[idx] + (1.0f - m) * xb[idx];
}

extern "C" void kernel_launch(void* const* d_in, const int* in_sizes, int n_in,
                              void* d_out, int out_size, void* d_ws,
                              size_t ws_size, hipStream_t stream) {
  const float* e0      = (const float*)d_in[0];
  const float* rel_emb = (const float*)d_in[1];
  const float* nW1     = (const float*)d_in[2];
  const float* nb1     = (const float*)d_in[3];
  const float* nW2     = (const float*)d_in[4];
  const float* nb2     = (const float*)d_in[5];
  const float* eW1     = (const float*)d_in[6];
  const float* eb1     = (const float*)d_in[7];
  const float* eW2     = (const float*)d_in[8];
  const float* eb2     = (const float*)d_in[9];
  const int*   adj_row = (const int*)d_in[10];
  const int*   adj_col = (const int*)d_in[11];
  const float* adj_val = (const float*)d_in[12];
  const int*   rw_row  = (const int*)d_in[13];
  const int*   rw_col  = (const int*)d_in[14];
  const float* rw_val  = (const float*)d_in[15];
  const float* node_eps = (const float*)d_in[16];
  const float* edge_eps = (const float*)d_in[17];
  const int*   sub     = (const int*)d_in[18];
  const int*   rel     = (const int*)d_in[19];

  const int E   = in_sizes[10];
  const int Erw = in_sizes[13];

  float* out    = (float*)d_out;
  float* scores = out;                                   // (1024, 50000)
  float* x_nd   = out + (size_t)BATCHB * N_ENTS;         // (50000, 200)
  float* x_ed   = x_nd + (size_t)N_ENTS * DIMF;          // (50000, 200)
  // scores region doubles as SpMM scratch (fully overwritten at the end)
  float* B1 = scores;
  float* B2 = scores + (size_t)N_ENTS * DIMF;

  // small workspace: rowptrs, masks, hr
  int* rp_adj = (int*)d_ws;                 // N+1 (padded to 50064)
  int* rp_rw  = rp_adj + 50064;
  float* nmask = (float*)(rp_rw + 50064);   // 2*N
  float* emask = nmask + 2 * N_ENTS;        // E (padded to 16)
  float* hr    = emask + (size_t)((E + 15) & ~15);  // 1024*200, 16B aligned

  dim3 b256(256);
  int rpBlocks = (N_ENTS + 1 + 255) / 256;
  rowptr_kernel<<<rpBlocks, b256, 0, stream>>>(adj_row, E, N_ENTS, rp_adj);
  rowptr_kernel<<<rpBlocks, b256, 0, stream>>>(rw_row, Erw, N_ENTS, rp_rw);

  // gates (WMMA)
  node_gate_kernel<<<dim3((N_ENTS + 63) / 64, 2), dim3(128), 0, stream>>>(
      e0, nW1, nb1, nW2, nb2, node_eps, nmask);
  edge_gate_kernel<<<dim3((E + 31) / 32), dim3(64), 0, stream>>>(
      e0, adj_row, adj_col, eW1 + (size_t)2 * DIMF * DIMF, eb1 + DIMF,
      eW2 + DIMF, eb2 + 1, edge_eps + E, emask, E);

  const int cmbBlocks = (int)(((size_t)N_ENTS * DIMF + 255) / 256);

  // edge-drop branch: x_ed = A(mask1) @ RW @ RW @ e0
  spmm_kernel<<<N_ENTS, b256, 0, stream>>>(rp_rw, rw_col, rw_val, nullptr, e0, B1);
  spmm_kernel<<<N_ENTS, b256, 0, stream>>>(rp_rw, rw_col, rw_val, nullptr, B1, B2);
  spmm_kernel<<<N_ENTS, b256, 0, stream>>>(rp_adj, adj_col, adj_val, emask, B2, x_ed);

  // node-drop branch (B1 still holds RW @ e0)
  combine_kernel<<<cmbBlocks, b256, 0, stream>>>(nmask, e0, B1, B2);          // x1
  spmm_kernel<<<N_ENTS, b256, 0, stream>>>(rp_adj, adj_col, adj_val, nullptr, B2, B1); // A x1
  spmm_kernel<<<N_ENTS, b256, 0, stream>>>(rp_rw, rw_col, rw_val, nullptr, B1, B2);    // mp1
  combine_kernel<<<cmbBlocks, b256, 0, stream>>>(nmask + N_ENTS, B1, B2, B2); // x2
  spmm_kernel<<<N_ENTS, b256, 0, stream>>>(rp_adj, adj_col, adj_val, nullptr, B2, x_nd);

  // scoring (WMMA), overwrites the scratch region last
  self_hr_kernel<<<BATCHB, b256, 0, stream>>>(e0, rel_emb, sub, rel, hr);
  scores_kernel<<<dim3((N_ENTS + 255) / 256, BATCHB / 64), dim3(128), 0, stream>>>(
      hr, e0, scores);
}
